// PhysicsNODE_78331613545136
// MI455X (gfx1250) — matrix operand order repaired and checked by reading, not compiled, not run on hardware
//
#include <hip/hip_runtime.h>

// ---------------------------------------------------------------------------
// Fused FiLM-conditioned neural-ODE (RK4) for MI455X / gfx1250.
// One persistent kernel: 512 blocks x 128 threads (4 wave32 waves).
// Each wave integrates a 16-sample tile with v_wmma_f32_16x16x32_f16 GEMMs.
// LDS footprint 146.7KB -> 2 blocks (8 waves) per 320KB WGP for co-execution.
// ---------------------------------------------------------------------------

typedef __attribute__((ext_vector_type(16))) _Float16 v16h;
typedef __attribute__((ext_vector_type(8)))  float    v8f;

#define NB        512          // blocks (32768 samples / 64 per block)
#define F_INIT    0.01f
#define F_EQ_TOL  0.999f
#define LN_EPS    1e-5f

// ---- LDS layout (bytes) ----------------------------------------------------
#define OFF_W0T    0           // ushort[128][64]   l0_w^T, K padded 44->64
#define OFF_W1T    16384       // ushort[128][128]  l1_w^T
#define OFF_WOT    49152       // ushort[16][128]   lout_w^T, N padded 9->16
#define OFF_L0B    53248       // float[128]
#define OFF_L1B    53760       // float[128]
#define OFF_LOB    54272       // float[16]
#define OFF_ATT_A  54336       // float[96]   Aq/Ak/Av
#define OFF_ATT_C  54720       // float[480]  Cq/Ck/Cv (bias folded)
#define OFF_FILM0  56640       // uint[64][128]  pack(f16(1+g0), f16(b0))
#define OFF_FILM1  89408       // uint[64][128]  pack(f16(1+g1), f16(b1))
#define OFF_X0     122176      // ushort[4][16][64]   x tile (y | cond | pad)
#define OFF_X1     130368      // ushort[4][16][128]  hidden act (reused L0/L1)
#define LDS_BYTES  146752

static __device__ __forceinline__ unsigned short f2h(float f) {
    _Float16 h = (_Float16)f;
    return __builtin_bit_cast(unsigned short, h);
}
static __device__ __forceinline__ float h2f(unsigned short u) {
    return (float)__builtin_bit_cast(_Float16, u);
}
static __device__ __forceinline__ unsigned pack2h(float a, float b) {
    return (unsigned)f2h(a) | ((unsigned)f2h(b) << 16);
}
// silu via hardware v_rcp_f32 (avoid IEEE divide chain)
static __device__ __forceinline__ float silu_fast(float x) {
    float e = __expf(-x);
    return x * __builtin_amdgcn_rcpf(1.f + e);
}

// Load a 32-byte WMMA operand fragment from LDS as two b128 chunks.
static __device__ __forceinline__ v16h ld16(const unsigned short* p0,
                                            const unsigned short* p1) {
    union { uint4 q[2]; v16h v; } u;
    u.q[0] = *(const uint4*)p0;
    u.q[1] = *(const uint4*)p1;
    return u.v;
}

extern "C" __global__ void __launch_bounds__(128)
node_fused_kernel(const float* __restrict__ sf,    // static_features [B][8]
                  const float* __restrict__ feqg,  // f_eq [B]
                  const float* __restrict__ ts,    // t_span [32]
                  const float* embed_w, const float* embed_b, const float* elem_embed,
                  const float* attn_in_w, const float* attn_in_b,
                  const float* attn_out_w, const float* attn_out_b,
                  const float* ln_g, const float* ln_b,
                  const float* proj_w, const float* proj_b,
                  const float* phys1_w, const float* phys1_b,
                  const float* phys2_w, const float* phys2_b,
                  const float* l0_w, const float* l0_b,
                  const float* l1_w, const float* l1_b,
                  const float* lout_w, const float* lout_b,
                  const float* f0gw, const float* f0gb,
                  const float* f0bw, const float* f0bb,
                  const float* f1gw, const float* f1gb,
                  const float* f1bw, const float* f1bb,
                  float* __restrict__ out)         // [B][32]
{
    extern __shared__ char smem[];
    unsigned short* W0t  = (unsigned short*)(smem + OFF_W0T);
    unsigned short* W1t  = (unsigned short*)(smem + OFF_W1T);
    unsigned short* Wot  = (unsigned short*)(smem + OFF_WOT);
    float* l0bl = (float*)(smem + OFF_L0B);
    float* l1bl = (float*)(smem + OFF_L1B);
    float* lobl = (float*)(smem + OFF_LOB);
    float* Al   = (float*)(smem + OFF_ATT_A);
    float* Cl   = (float*)(smem + OFF_ATT_C);
    unsigned* film0 = (unsigned*)(smem + OFF_FILM0);
    unsigned* film1 = (unsigned*)(smem + OFF_FILM1);
    unsigned short* x0 = (unsigned short*)(smem + OFF_X0);
    unsigned short* x1 = (unsigned short*)(smem + OFF_X1);

    const int tid = threadIdx.x;

    // ---------------- Phase 1: stage weights into LDS (transposed f16) -----
    for (int t = tid; t < 128 * 64; t += 128) {      // W0t[n][k] = l0_w[k][n]
        int n = t >> 6, k = t & 63;
        W0t[t] = (k < 44) ? f2h(l0_w[k * 128 + n]) : (unsigned short)0;
    }
    for (int t = tid; t < 128 * 128; t += 128) {     // W1t[n][k] = l1_w[k][n]
        int n = t >> 7, k = t & 127;
        W1t[t] = f2h(l1_w[k * 128 + n]);
    }
    for (int t = tid; t < 16 * 128; t += 128) {      // Wot[n][k] = lout_w[k][n]
        int n = t >> 7, k = t & 127;
        Wot[t] = (n < 9) ? f2h(lout_w[k * 9 + n]) : (unsigned short)0;
    }
    if (tid < 128) { l0bl[tid] = l0_b[tid]; l1bl[tid] = l1_b[tid]; }
    if (tid < 16)  lobl[tid] = (tid < 9) ? lout_b[tid] : 0.f;

    // Attention affine constants: tokens[i] = comp[i]*embed_w + (embed_b+elem_embed[i])
    // => q/k/v[i] = comp[i]*A + C[i]  (bias folded into C).
    for (int t = tid; t < 96; t += 128) {
        int which = t >> 5, e2 = t & 31, cw = which * 32 + e2;
        float a = 0.f;
        for (int e = 0; e < 32; e++) a += embed_w[e] * attn_in_w[e * 96 + cw];
        Al[t] = a;
    }
    for (int t = tid; t < 480; t += 128) {
        int which = t / 160, rem = t % 160, i = rem >> 5, e2 = rem & 31;
        int cw = which * 32 + e2;
        float c = attn_in_b[cw];
        for (int e = 0; e < 32; e++)
            c += (embed_b[e] + elem_embed[i * 32 + e]) * attn_in_w[e * 96 + cw];
        Cl[t] = c;
    }
    __syncthreads();

    // ---------------- Phase 2: conditioning + FiLM (64 threads) ------------
    if (tid < 64) {
        const int ls = tid;
        const int gs = blockIdx.x * 64 + ls;
        float comp[5];
        #pragma unroll
        for (int i = 0; i < 5; i++) comp[i] = sf[gs * 8 + i];

        float obar[32];
        #pragma unroll
        for (int e = 0; e < 32; e++) obar[e] = 0.f;

        const float* Aq = Al;       const float* Ak = Al + 32;  const float* Av = Al + 64;
        const float* Cq = Cl;       const float* Ck = Cl + 160; const float* Cv = Cl + 320;

        for (int hh = 0; hh < 4; hh++) {             // 4 heads, dh = 8
            float qh[5][8], kh[5][8], vh[5][8];
            #pragma unroll
            for (int i = 0; i < 5; i++) {
                #pragma unroll
                for (int d = 0; d < 8; d++) {
                    int e2 = hh * 8 + d;
                    qh[i][d] = comp[i] * Aq[e2] + Cq[i * 32 + e2];
                    kh[i][d] = comp[i] * Ak[e2] + Ck[i * 32 + e2];
                    vh[i][d] = comp[i] * Av[e2] + Cv[i * 32 + e2];
                }
            }
            #pragma unroll
            for (int i = 0; i < 5; i++) {
                float sc[5]; float mx = -1e30f;
                #pragma unroll
                for (int j = 0; j < 5; j++) {
                    float s = 0.f;
                    #pragma unroll
                    for (int d = 0; d < 8; d++) s += qh[i][d] * kh[j][d];
                    sc[j] = s * 0.35355339059327f;   // 1/sqrt(8)
                    mx = fmaxf(mx, sc[j]);
                }
                float den = 0.f;
                #pragma unroll
                for (int j = 0; j < 5; j++) { sc[j] = __expf(sc[j] - mx); den += sc[j]; }
                float inv = 0.2f * __builtin_amdgcn_rcpf(den);  // softmax * mean-over-i
                #pragma unroll
                for (int d = 0; d < 8; d++) {
                    float o = 0.f;
                    #pragma unroll
                    for (int j = 0; j < 5; j++) o += sc[j] * vh[j][d];
                    obar[hh * 8 + d] += o * inv;
                }
            }
        }
        // out-proj (after mean, linear commutes), layernorm, proj
        float pooled[32]; float mu = 0.f;
        #pragma unroll
        for (int e2 = 0; e2 < 32; e2++) {
            float p = attn_out_b[e2];
            #pragma unroll
            for (int e = 0; e < 32; e++) p += obar[e] * attn_out_w[e * 32 + e2];
            pooled[e2] = p; mu += p;
        }
        mu *= (1.f / 32.f);
        float var = 0.f;
        #pragma unroll
        for (int e2 = 0; e2 < 32; e2++) { float d = pooled[e2] - mu; var += d * d; }
        float rs = rsqrtf(var * (1.f / 32.f) + LN_EPS);
        float normed[32];
        #pragma unroll
        for (int e2 = 0; e2 < 32; e2++)
            normed[e2] = (pooled[e2] - mu) * rs * ln_g[e2] + ln_b[e2];
        float cond[35];
        #pragma unroll
        for (int e2 = 0; e2 < 32; e2++) {
            float c = proj_b[e2];
            #pragma unroll
            for (int e = 0; e < 32; e++) c += normed[e] * proj_w[e * 32 + e2];
            cond[e2] = c;
        }
        // physics MLP 3 -> 32 (silu) -> 3
        float pf0 = sf[gs * 8 + 5], pf1 = sf[gs * 8 + 6], pf2 = sf[gs * 8 + 7];
        float hs[32];
        #pragma unroll
        for (int e = 0; e < 32; e++) {
            float v = phys1_b[e] + pf0 * phys1_w[e] + pf1 * phys1_w[32 + e] + pf2 * phys1_w[64 + e];
            hs[e] = silu_fast(v);
        }
        #pragma unroll
        for (int c = 0; c < 3; c++) {
            float o = phys2_b[c];
            #pragma unroll
            for (int e = 0; e < 32; e++) o += hs[e] * phys2_w[e * 3 + c];
            cond[32 + c] = o;
        }
        // write cond into x-tile cols 9..43, zero the K pad 44..63
        unsigned short* xr = x0 + (ls >> 4) * (16 * 64) + (ls & 15) * 64;
        #pragma unroll
        for (int j = 0; j < 35; j++) xr[9 + j] = f2h(cond[j]);
        #pragma unroll
        for (int c = 44; c < 64; c++) xr[c] = (unsigned short)0;
        // FiLM params into LDS: pack(1+gamma, beta) as 2xf16
        for (int n = 0; n < 128; n++) {
            float a0 = f0gb[n], a1 = f0bb[n], a2 = f1gb[n], a3 = f1bb[n];
            #pragma unroll
            for (int k = 0; k < 35; k++) {
                float c = cond[k];
                a0 += c * f0gw[k * 128 + n];
                a1 += c * f0bw[k * 128 + n];
                a2 += c * f1gw[k * 128 + n];
                a3 += c * f1bw[k * 128 + n];
            }
            film0[ls * 128 + n] = pack2h(1.f + a0, a1);
            film1[ls * 128 + n] = pack2h(1.f + a2, a3);
        }
        // trajectory t=0
        out[gs * 32 + 0] = fminf(F_INIT, feqg[gs] * F_EQ_TOL);
    }
    __syncthreads();

    // ---------------- Phase 3: RK4 integration with WMMA --------------------
    const int lane = tid & 31;
    const int w    = tid >> 5;        // wave id: owns 16-sample tile
    const int col  = lane & 15;       // N column / row-M selector
    const int hi   = lane >> 4;       // lane half
    const int tb   = blockIdx.x * 64 + w * 16;

    unsigned short* x0w = x0 + w * (16 * 64);
    unsigned short* x1w = x1 + w * (16 * 128);

    const v8f vzero = {0.f, 0.f, 0.f, 0.f, 0.f, 0.f, 0.f, 0.f};

    float yv[8], kp[8], feq[8];
    #pragma unroll
    for (int r = 0; r < 8; r++) {
        yv[r] = (col == 0) ? F_INIT : 0.f;
        kp[r] = 0.f;
        feq[r] = 1e30f;
    }
    if (col == 0) {
        #pragma unroll
        for (int r = 0; r < 8; r++) feq[r] = feqg[tb + r + 8 * hi] * F_EQ_TOL;
    }

    for (int i = 0; i < 31; i++) {
        float h = ts[i + 1] - ts[i];
        float ynew[8];
        #pragma unroll
        for (int r = 0; r < 8; r++) ynew[r] = yv[r];

        for (int s4 = 0; s4 < 4; s4++) {
            float prr = (s4 == 0) ? 0.f : ((s4 == 3) ? h : 0.5f * h);
            float wgt = (s4 == 1 || s4 == 2) ? h * (1.f / 3.f) : h * (1.f / 6.f);

            // stage input y+prr*k into x-tile cols 0..8 (f16)
            if (col < 9) {
                #pragma unroll
                for (int r = 0; r < 8; r++)
                    x0w[(r + 8 * hi) * 64 + col] = f2h(yv[r] + prr * kp[r]);
            }
            asm volatile("" ::: "memory");

            // ---- layer 0: (16x64) @ (64x128) -------------------------------
            v8f acc[8];
            #pragma unroll
            for (int nt = 0; nt < 8; nt++) acc[nt] = vzero;
            #pragma unroll
            for (int kt = 0; kt < 2; kt++) {
                const unsigned short* ar = x0w + col * 64 + kt * 32 + hi * 8;
                v16h A = ld16(ar, ar + 16);
                #pragma unroll
                for (int nt = 0; nt < 8; nt++) {
                    const unsigned short* br = W0t + (nt * 16 + col) * 64 + kt * 32 + hi * 8;
                    v16h Bo = ld16(br, br + 16);
                    acc[nt] = __builtin_amdgcn_wmma_f32_16x16x32_f16(
                        false, A, false, Bo, (short)0, acc[nt], false, false);
                }
            }
            // bias + silu + FiLM0 -> x1 (f16)
            #pragma unroll
            for (int nt = 0; nt < 8; nt++) {
                int n = nt * 16 + col;
                float bia = l0bl[n];
                #pragma unroll
                for (int r = 0; r < 8; r++) {
                    int lsamp = w * 16 + r + 8 * hi;
                    unsigned fp = film0[lsamp * 128 + n];
                    float v = silu_fast(acc[nt][r] + bia);
                    v = v * h2f((unsigned short)fp) + h2f((unsigned short)(fp >> 16));
                    x1w[(r + 8 * hi) * 128 + n] = f2h(v);
                }
            }
            asm volatile("" ::: "memory");

            // ---- layer 1: (16x128) @ (128x128) -----------------------------
            #pragma unroll
            for (int nt = 0; nt < 8; nt++) acc[nt] = vzero;
            #pragma unroll
            for (int kt = 0; kt < 4; kt++) {
                const unsigned short* ar = x1w + col * 128 + kt * 32 + hi * 8;
                v16h A = ld16(ar, ar + 16);
                #pragma unroll
                for (int nt = 0; nt < 8; nt++) {
                    const unsigned short* br = W1t + (nt * 16 + col) * 128 + kt * 32 + hi * 8;
                    v16h Bo = ld16(br, br + 16);
                    acc[nt] = __builtin_amdgcn_wmma_f32_16x16x32_f16(
                        false, A, false, Bo, (short)0, acc[nt], false, false);
                }
            }
            // bias + silu + FiLM1 -> x1 (safe: all A-frags consumed via the
            // accumulator data dependency before these stores issue)
            #pragma unroll
            for (int nt = 0; nt < 8; nt++) {
                int n = nt * 16 + col;
                float bia = l1bl[n];
                #pragma unroll
                for (int r = 0; r < 8; r++) {
                    int lsamp = w * 16 + r + 8 * hi;
                    unsigned fp = film1[lsamp * 128 + n];
                    float v = silu_fast(acc[nt][r] + bia);
                    v = v * h2f((unsigned short)fp) + h2f((unsigned short)(fp >> 16));
                    x1w[(r + 8 * hi) * 128 + n] = f2h(v);
                }
            }
            asm volatile("" ::: "memory");

            // ---- output layer: (16x128) @ (128x16, 9 live cols) ------------
            v8f ak = vzero;
            #pragma unroll
            for (int kt = 0; kt < 4; kt++) {
                const unsigned short* ar = x1w + col * 128 + kt * 32 + hi * 8;
                v16h A = ld16(ar, ar + 16);
                const unsigned short* br = Wot + col * 128 + kt * 32 + hi * 8;
                v16h Bo = ld16(br, br + 16);
                ak = __builtin_amdgcn_wmma_f32_16x16x32_f16(
                    false, A, false, Bo, (short)0, ak, false, false);
            }
            float lb = lobl[col];
            #pragma unroll
            for (int r = 0; r < 8; r++) {
                kp[r] = ak[r] + lb;               // k_s
                ynew[r] += wgt * kp[r];           // RK4 accumulation
            }
        }
        #pragma unroll
        for (int r = 0; r < 8; r++) yv[r] = ynew[r];

        if (col == 0) {
            #pragma unroll
            for (int r = 0; r < 8; r++)
                out[(tb + r + 8 * hi) * 32 + i + 1] =
                    fminf(fmaxf(yv[r], 0.f), feq[r]);
        }
    }
}

extern "C" void kernel_launch(void* const* d_in, const int* in_sizes, int n_in,
                              void* d_out, int out_size, void* d_ws, size_t ws_size,
                              hipStream_t stream) {
    (void)in_sizes; (void)n_in; (void)d_ws; (void)ws_size; (void)out_size;
    const float* sf        = (const float*)d_in[0];   // static_features
    const float* feq       = (const float*)d_in[1];   // f_eq
    /* d_in[2] = delta_G_norm, unused by reference */
    const float* ts        = (const float*)d_in[3];   // t_span
    const float* embed_w   = (const float*)d_in[4];
    const float* embed_b   = (const float*)d_in[5];
    const float* elem_embed= (const float*)d_in[6];
    const float* attn_in_w = (const float*)d_in[7];
    const float* attn_in_b = (const float*)d_in[8];
    const float* attn_out_w= (const float*)d_in[9];
    const float* attn_out_b= (const float*)d_in[10];
    const float* ln_g      = (const float*)d_in[11];
    const float* ln_b      = (const float*)d_in[12];
    const float* proj_w    = (const float*)d_in[13];
    const float* proj_b    = (const float*)d_in[14];
    const float* phys1_w   = (const float*)d_in[15];
    const float* phys1_b   = (const float*)d_in[16];
    const float* phys2_w   = (const float*)d_in[17];
    const float* phys2_b   = (const float*)d_in[18];
    const float* l0_w      = (const float*)d_in[19];
    const float* l0_b      = (const float*)d_in[20];
    const float* l1_w      = (const float*)d_in[21];
    const float* l1_b      = (const float*)d_in[22];
    const float* lout_w    = (const float*)d_in[23];
    const float* lout_b    = (const float*)d_in[24];
    const float* f0gw      = (const float*)d_in[25];
    const float* f0gb      = (const float*)d_in[26];
    const float* f0bw      = (const float*)d_in[27];
    const float* f0bb      = (const float*)d_in[28];
    const float* f1gw      = (const float*)d_in[29];
    const float* f1gb      = (const float*)d_in[30];
    const float* f1bw      = (const float*)d_in[31];
    const float* f1bb      = (const float*)d_in[32];
    float* out = (float*)d_out;

    (void)hipFuncSetAttribute(reinterpret_cast<const void*>(node_fused_kernel),
                              hipFuncAttributeMaxDynamicSharedMemorySize, LDS_BYTES);

    node_fused_kernel<<<dim3(NB), dim3(128), LDS_BYTES, stream>>>(
        sf, feq, ts,
        embed_w, embed_b, elem_embed,
        attn_in_w, attn_in_b, attn_out_w, attn_out_b,
        ln_g, ln_b, proj_w, proj_b,
        phys1_w, phys1_b, phys2_w, phys2_b,
        l0_w, l0_b, l1_w, l1_b, lout_w, lout_b,
        f0gw, f0gb, f0bw, f0bb, f1gw, f1gb, f1bw, f1bb,
        out);
}
